// Attention_8495445311883
// MI455X (gfx1250) — matrix operand order repaired and checked by reading, not compiled, not run on hardware
//
#include <hip/hip_runtime.h>

// Problem constants
#define CV 10000
#define CH 512
#define CT 128
#define CB 32
#define CM 4096   // T*B

typedef __bf16 bf16_t;
typedef __attribute__((ext_vector_type(8)))  bf16_t v8bf;
typedef __attribute__((ext_vector_type(16))) bf16_t v16bf;
typedef __attribute__((ext_vector_type(8)))  float  v8f;

// ---------------------------------------------------------------------------
// WMMA GEMM:  C = act( A @ W^T + bias [+ badd(m%B, :)] )
//   A   : f32, logical [M x K]; physical row via amode:
//         0: phys=m   1: phys=(m%B)*T + m/B   2: phys=(m%B)*T + (T-1 - m/B)
//   W   : f32 [N x K] row-major (PyTorch-style weight)
//   C   : f32, physical row via cmode (0: m, 1: (m%B)*T + m/B), stride ldc
//   badd: optional f32 [B x N] broadcast over time (row = m % B)
//   act : 0 = none, 1 = tanh
// Tile: 128x128x32, 256 threads = 8 waves (4 x 2), each wave 2x4 16x16 frags.
// Double-buffered LDS staging (f32 -> bf16 convert on the fly), hoisted
// addressing, K-tail split out, global prefetch two tiles ahead.
// ---------------------------------------------------------------------------
__global__ __launch_bounds__(256)
void gemm_wmma_bf16(const float* __restrict__ A, int amode, int lda,
                    const float* __restrict__ W,
                    const float* __restrict__ bias,
                    const float* __restrict__ badd,
                    float* __restrict__ C, int cmode, int ldc,
                    int M, int N, int K, int act)
{
    constexpr int BM = 128, BN = 128, BK = 32, LSTR = 40; // LDS stride (bf16)
    __shared__ __align__(16) bf16_t As[2][BM * LSTR];
    __shared__ __align__(16) bf16_t Bs[2][BN * LSTR];

    const int tid  = threadIdx.x;
    const int lane = tid & 31;
    const int wave = tid >> 5;
    const int wm = wave & 3;          // 0..3 -> 32-row strip
    const int wn = wave >> 2;         // 0..1 -> 64-col strip
    const int half = lane >> 4;       // K-half selector (ISA bf16 layout)
    const int r16  = lane & 15;

    const int n0 = blockIdx.x * BN;
    const int m0 = blockIdx.y * BM;

    // ---- loop-invariant staging descriptors (4 slots per thread each for A/B)
    const float* abase[4];  int aoff[4]; int akq[4]; bool amv[4];
    const float* bbase[4];  int boff[4]; int bkq[4]; bool bnv[4];
    #pragma unroll
    for (int it = 0; it < 4; ++it) {
        int s   = it * 256 + tid;
        int row = s >> 3;
        int kq  = (s & 7) << 2;
        int m   = m0 + row;
        int phys = (amode == 0) ? m
                 : (amode == 1) ? ((m & (CB - 1)) * CT + (m >> 5))
                 :                ((m & (CB - 1)) * CT + (CT - 1 - (m >> 5)));
        amv[it]   = (m < M);
        abase[it] = A + (size_t)phys * (size_t)lda + kq;
        aoff[it]  = row * LSTR + kq;
        akq[it]   = kq;
        int n     = n0 + row;
        bnv[it]   = (n < N);
        bbase[it] = W + (size_t)n * (size_t)K + kq;
        boff[it]  = row * LSTR + kq;
        bkq[it]   = kq;
    }
    // loop-invariant fragment LDS offsets
    int afoff[2], bfoff[4];
    #pragma unroll
    for (int mi = 0; mi < 2; ++mi)
        afoff[mi] = (wm * 32 + mi * 16 + r16) * LSTR + half * 8;
    #pragma unroll
    for (int ni = 0; ni < 4; ++ni)
        bfoff[ni] = (wn * 64 + ni * 16 + r16) * LSTR + half * 8;

    v8f acc[2][4] = {};

    const int ktiles = (K + BK - 1) / BK;

    // fetch one k-tile worth into registers (full: unguarded vector loads)
    auto fetch = [&](int k0, bool full, float4* ra, float4* rb) {
        #pragma unroll
        for (int it = 0; it < 4; ++it) {
            float4 f = make_float4(0.f, 0.f, 0.f, 0.f);
            if (amv[it]) {
                if (full) {
                    f = *(const float4*)(abase[it] + k0);
                } else {
                    const float* p = abase[it] + k0;
                    int kk = k0 + akq[it];
                    if (kk + 0 < K) f.x = p[0];
                    if (kk + 1 < K) f.y = p[1];
                    if (kk + 2 < K) f.z = p[2];
                    if (kk + 3 < K) f.w = p[3];
                }
            }
            ra[it] = f;
        }
        #pragma unroll
        for (int it = 0; it < 4; ++it) {
            float4 f = make_float4(0.f, 0.f, 0.f, 0.f);
            if (bnv[it]) {
                if (full) {
                    f = *(const float4*)(bbase[it] + k0);
                } else {
                    const float* p = bbase[it] + k0;
                    int kk = k0 + bkq[it];
                    if (kk + 0 < K) f.x = p[0];
                    if (kk + 1 < K) f.y = p[1];
                    if (kk + 2 < K) f.z = p[2];
                    if (kk + 3 < K) f.w = p[3];
                }
            }
            rb[it] = f;
        }
    };
    // convert + store registers into LDS buffer `bf`
    auto stash = [&](int bfi, const float4* ra, const float4* rb) {
        #pragma unroll
        for (int it = 0; it < 4; ++it) {
            bf16_t* p = &As[bfi][aoff[it]];
            p[0] = (bf16_t)ra[it].x; p[1] = (bf16_t)ra[it].y;
            p[2] = (bf16_t)ra[it].z; p[3] = (bf16_t)ra[it].w;
            bf16_t* q = &Bs[bfi][boff[it]];
            q[0] = (bf16_t)rb[it].x; q[1] = (bf16_t)rb[it].y;
            q[2] = (bf16_t)rb[it].z; q[3] = (bf16_t)rb[it].w;
        }
    };

    // prologue: stage tile 0 into buffer 0
    {
        float4 ra[4], rb[4];
        fetch(0, BK <= K, ra, rb);
        stash(0, ra, rb);
    }
    int buf = 0;
    for (int kt = 0; kt < ktiles; ++kt) {
        __syncthreads();
        const bool have_next = (kt + 1 < ktiles);
        float4 ra[4], rb[4];
        if (have_next)
            fetch((kt + 1) * BK, (kt + 2) * BK <= K, ra, rb);
        if (kt + 2 < ktiles) {
            const int pk = (kt + 2) * BK;
            #pragma unroll
            for (int it = 0; it < 4; ++it) {
                __builtin_prefetch(abase[it] + pk, 0, 1);
                __builtin_prefetch(bbase[it] + pk, 0, 1);
            }
        }

        // ---- fragment loads per ISA bf16 layouts ----
        // A 16x32: lane<16 -> K {0..7,16..23}, lane>=16 -> K {8..15,24..31}
        v16bf afrag[2], bfrag[4];
        #pragma unroll
        for (int mi = 0; mi < 2; ++mi) {
            v8bf lo = *(const v8bf*)&As[buf][afoff[mi]];
            v8bf hi = *(const v8bf*)&As[buf][afoff[mi] + 16];
            #pragma unroll
            for (int e = 0; e < 8; ++e) { afrag[mi][e] = lo[e]; afrag[mi][e + 8] = hi[e]; }
        }
        // B 32x16: lane<16 -> col=lane, K {0..7,16..23} ; lane>=16 -> K {8..15,24..31}
        #pragma unroll
        for (int ni = 0; ni < 4; ++ni) {
            v8bf lo = *(const v8bf*)&Bs[buf][bfoff[ni]];
            v8bf hi = *(const v8bf*)&Bs[buf][bfoff[ni] + 16];
            #pragma unroll
            for (int e = 0; e < 8; ++e) { bfrag[ni][e] = lo[e]; bfrag[ni][e + 8] = hi[e]; }
        }
        #pragma unroll
        for (int mi = 0; mi < 2; ++mi)
            #pragma unroll
            for (int ni = 0; ni < 4; ++ni)
                acc[mi][ni] = __builtin_amdgcn_wmma_f32_16x16x32_bf16(
                    false, afrag[mi], false, bfrag[ni],
                    (short)0, acc[mi][ni], false, false);

        if (have_next)
            stash(buf ^ 1, ra, rb);
        buf ^= 1;
    }

    // ---- epilogue: C/D f32 layout (VGPR r, lane -> row r + 8*(lane/16), col lane%16)
    #pragma unroll
    for (int mi = 0; mi < 2; ++mi) {
        #pragma unroll
        for (int ni = 0; ni < 4; ++ni) {
            int col = wn * 64 + ni * 16 + r16;
            int n = n0 + col;
            if (n >= N) continue;
            #pragma unroll
            for (int r = 0; r < 8; ++r) {
                int row = wm * 32 + mi * 16 + half * 8 + r;
                int m = m0 + row;
                if (m >= M) continue;
                float v = acc[mi][ni][r];
                if (bias) v += bias[n];
                if (badd) v += badd[(m & (CB - 1)) * N + n];
                if (act)  v = tanhf(v);
                int phys = (cmode == 0) ? m : ((m & (CB - 1)) * CT + (m >> 5));
                C[(size_t)phys * (size_t)ldc + n] = v;
            }
        }
    }
}

// ---------------------------------------------------------------------------
// One RNN time-step, both directions. grid = (B*2, 2), block = 256.
// enc_cat is [T,B,2H]; forward states in cols [0,H), backward in [H,2H)
// stored at forward-time positions. Previous state read from enc_cat.
// ---------------------------------------------------------------------------
__global__ __launch_bounds__(256)
void rnn_step(const float* __restrict__ pre_f, const float* __restrict__ pre_b,
              const float* __restrict__ Whh_f, const float* __restrict__ bhh_f,
              const float* __restrict__ Whh_b, const float* __restrict__ bhh_b,
              const float* __restrict__ hinit,     // [2,B,H]
              float* __restrict__ enc_cat, int t)
{
    __shared__ float hs[CH];
    const int dir  = blockIdx.y;
    const int b    = blockIdx.x >> 1;
    const int hhalf = blockIdx.x & 1;
    const int n    = hhalf * 256 + threadIdx.x;

    const float* pre  = dir ? pre_b : pre_f;
    const float* Whh  = dir ? Whh_b : Whh_f;
    const float* bhh  = dir ? bhh_b : bhh_f;

    const float* hp;
    if (dir == 0) {
        hp = (t == 0) ? (hinit + (size_t)b * CH)
                      : (enc_cat + ((size_t)(t - 1) * CB + b) * (2 * CH));
    } else {
        int tt = CT - 1 - t;
        hp = (t == 0) ? (hinit + (size_t)(CB + b) * CH)
                      : (enc_cat + ((size_t)(tt + 1) * CB + b) * (2 * CH) + CH);
    }
    hs[threadIdx.x]       = hp[threadIdx.x];
    hs[threadIdx.x + 256] = hp[threadIdx.x + 256];
    __syncthreads();

    float acc = pre[((size_t)t * CB + b) * CH + n] + bhh[n];
    const float* wr = Whh + (size_t)n * CH;
    #pragma unroll 4
    for (int k = 0; k < CH; k += 4) {
        float4 w = *(const float4*)(wr + k);
        acc += hs[k] * w.x + hs[k + 1] * w.y + hs[k + 2] * w.z + hs[k + 3] * w.w;
    }
    float out = tanhf(acc);
    int tpos = dir ? (CT - 1 - t) : t;
    enc_cat[((size_t)tpos * CB + b) * (2 * CH) + dir * CH + n] = out;
}

// ---------------------------------------------------------------------------
// h0 = cat(h_f, h_b) @ W_attn1^T + b_attn1     (h_f, h_b pulled from enc_cat)
// grid = 64 x 256
// ---------------------------------------------------------------------------
__global__ __launch_bounds__(256)
void k_h0(const float* __restrict__ enc_cat, const float* __restrict__ W_attn1,
          const float* __restrict__ b_attn1, float* __restrict__ h0)
{
    int idx = blockIdx.x * 256 + threadIdx.x;
    int b = idx >> 9, n = idx & 511;
    const float* hf = enc_cat + ((size_t)(CT - 1) * CB + b) * (2 * CH);       // fwd @ t=T-1
    const float* hb = enc_cat + (size_t)b * (2 * CH) + CH;                    // bwd final @ t=0
    const float* wr = W_attn1 + (size_t)n * (2 * CH);
    float acc = b_attn1[n];
    for (int k = 0; k < CH; ++k) acc += hf[k] * wr[k];
    for (int k = 0; k < CH; ++k) acc += hb[k] * wr[CH + k];
    h0[(size_t)b * CH + n] = acc;
}

// hp = h0 @ Whh_d^T + bhh_d
__global__ __launch_bounds__(256)
void k_hp(const float* __restrict__ h0, const float* __restrict__ Whh_d,
          const float* __restrict__ bhh_d, float* __restrict__ hpv)
{
    int idx = blockIdx.x * 256 + threadIdx.x;
    int b = idx >> 9, n = idx & 511;
    const float* hr = h0 + (size_t)b * CH;
    const float* wr = Whh_d + (size_t)n * CH;
    float acc = bhh_d[n];
    for (int k = 0; k < CH; ++k) acc += hr[k] * wr[k];
    hpv[(size_t)b * CH + n] = acc;
}

// ---------------------------------------------------------------------------
// Attention for one (i, b): scores over j (T=128), softmax, context.
// dec_out lives in cat2[:, H:2H]; context written to cat2[:, 0:H].
// grid = (T, B), block = 128.
// ---------------------------------------------------------------------------
__global__ __launch_bounds__(128)
void k_attn(const float* __restrict__ enc_out,   // [T*B, H]
            float* __restrict__ cat2)            // [T*B, 2H]
{
    __shared__ float dvec[CH];
    __shared__ float sc[CT];
    __shared__ float red[CT];
    const int i = blockIdx.x, b = blockIdx.y, tid = threadIdx.x;
    const size_t mrow = (size_t)i * CB + b;

    // load decoder vector
    float4 dv = *(const float4*)(cat2 + mrow * (2 * CH) + CH + tid * 4);
    *(float4*)(dvec + tid * 4) = dv;
    __syncthreads();

    // score for j = tid
    const float* er = enc_out + ((size_t)tid * CB + b) * CH;
    float acc = 0.f;
    #pragma unroll 4
    for (int h = 0; h < CH; ++h) acc += dvec[h] * er[h];
    sc[tid] = acc;
    red[tid] = acc;
    __syncthreads();
    // max-reduce
    for (int off = 64; off > 0; off >>= 1) {
        if (tid < off) red[tid] = fmaxf(red[tid], red[tid + off]);
        __syncthreads();
    }
    float mx = red[0];
    __syncthreads();
    float e = expf(sc[tid] - mx);
    sc[tid] = e;
    red[tid] = e;
    __syncthreads();
    for (int off = 64; off > 0; off >>= 1) {
        if (tid < off) red[tid] += red[tid + off];
        __syncthreads();
    }
    float inv = 1.0f / red[0];
    __syncthreads();
    sc[tid] *= inv;
    __syncthreads();

    // context: each thread owns 4 h-values
    int hb = tid * 4;
    float4 ctx = make_float4(0.f, 0.f, 0.f, 0.f);
    for (int j = 0; j < CT; ++j) {
        float wj = sc[j];
        float4 ev = *(const float4*)(enc_out + ((size_t)j * CB + b) * CH + hb);
        ctx.x += wj * ev.x; ctx.y += wj * ev.y;
        ctx.z += wj * ev.z; ctx.w += wj * ev.w;
    }
    *(float4*)(cat2 + mrow * (2 * CH) + hb) = ctx;
}

// ---------------------------------------------------------------------------
extern "C" void kernel_launch(void* const* d_in, const int* in_sizes, int n_in,
                              void* d_out, int out_size, void* d_ws, size_t ws_size,
                              hipStream_t stream)
{
    (void)in_sizes; (void)n_in; (void)out_size; (void)ws_size;
    const float* enc_inputs = (const float*)d_in[0];
    const float* enc_hidden = (const float*)d_in[1];
    const float* dec_inputs = (const float*)d_in[2];
    const float* Wih_f = (const float*)d_in[3];
    const float* Whh_f = (const float*)d_in[4];
    const float* bih_f = (const float*)d_in[5];
    const float* bhh_f = (const float*)d_in[6];
    const float* Wih_b = (const float*)d_in[7];
    const float* Whh_b = (const float*)d_in[8];
    const float* bih_b = (const float*)d_in[9];
    const float* bhh_b = (const float*)d_in[10];
    const float* Wih_d = (const float*)d_in[11];
    const float* Whh_d = (const float*)d_in[12];
    const float* bih_d = (const float*)d_in[13];
    const float* bhh_d = (const float*)d_in[14];
    const float* W_attn1 = (const float*)d_in[15];
    const float* b_attn1 = (const float*)d_in[16];
    const float* W_attn2 = (const float*)d_in[17];
    const float* b_attn2 = (const float*)d_in[18];
    const float* W_out = (const float*)d_in[19];
    const float* b_out = (const float*)d_in[20];
    float* out = (float*)d_out;

    // workspace carve-up (floats)
    float* pre_f   = (float*)d_ws;                 // [4096, 512]
    float* pre_b   = pre_f + (size_t)CM * CH;      // [4096, 512]
    float* enc_cat = pre_b + (size_t)CM * CH;      // [4096, 1024]
    float* enc_out = enc_cat + (size_t)CM * 2 * CH;// [4096, 512]
    float* cat2    = enc_out + (size_t)CM * CH;    // [4096, 1024]  (ctx | dec_out)
    float* h0b     = cat2 + (size_t)CM * 2 * CH;   // [32, 512]
    float* hpb     = h0b + (size_t)CB * CH;        // [32, 512]

    dim3 blk(256);
    dim3 gin(CH / 128, CM / 128);                  // N=512 tiles x M tiles

    // 1) pre_f = x @ Wih_f^T + bih_f   (x gathered from [B,T,V], time-forward)
    gemm_wmma_bf16<<<gin, blk, 0, stream>>>(enc_inputs, 1, CV, Wih_f, bih_f, nullptr,
                                            pre_f, 0, CH, CM, CH, CV, 0);
    // 2) pre_b (time-reversed gather)
    gemm_wmma_bf16<<<gin, blk, 0, stream>>>(enc_inputs, 2, CV, Wih_b, bih_b, nullptr,
                                            pre_b, 0, CH, CM, CH, CV, 0);
    // 3) bidirectional RNN scan, 128 sequential steps, both dirs per launch
    for (int t = 0; t < CT; ++t) {
        rnn_step<<<dim3(CB * 2, 2), blk, 0, stream>>>(pre_f, pre_b, Whh_f, bhh_f,
                                                      Whh_b, bhh_b, enc_hidden,
                                                      enc_cat, t);
    }
    // 4) h0 = cat(h_f,h_b) @ W_attn1^T + b_attn1 ; 5) hp = h0 @ Whh_d^T + bhh_d
    k_h0<<<dim3(64), blk, 0, stream>>>(enc_cat, W_attn1, b_attn1, h0b);
    k_hp<<<dim3(64), blk, 0, stream>>>(h0b, Whh_d, bhh_d, hpb);
    // 6) dec_out = tanh(dx @ Wih_d^T + bih_d + hp)  -> cat2[:, H:2H]
    gemm_wmma_bf16<<<gin, blk, 0, stream>>>(dec_inputs, 1, CV, Wih_d, bih_d, hpb,
                                            cat2 + CH, 0, 2 * CH, CM, CH, CV, 1);
    // 7) enc_out = enc_cat @ W_attn2^T + b_attn2
    gemm_wmma_bf16<<<gin, blk, 0, stream>>>(enc_cat, 0, 2 * CH, W_attn2, b_attn2, nullptr,
                                            enc_out, 0, CH, CM, CH, 2 * CH, 0);
    // 8) attention: scores/softmax/context -> cat2[:, 0:H]
    k_attn<<<dim3(CT, CB), dim3(128), 0, stream>>>(enc_out, cat2);
    // 9) predict = cat2 @ W_out^T + b_out  -> d_out as [B,T,V]
    gemm_wmma_bf16<<<dim3((CV + 127) / 128, CM / 128), blk, 0, stream>>>(
        cat2, 0, 2 * CH, W_out, b_out, nullptr,
        out, 1, CV, CM, CV, 2 * CH, 0);
}
